// FullModelSnowflake_84413287236175
// MI455X (gfx1250) — matrix operand order repaired and checked by reading, not compiled, not run on hardware
//
#include <hip/hip_runtime.h>
#include <hip/hip_bf16.h>
#include <math.h>

typedef __attribute__((ext_vector_type(16))) _Float16 v16h;
typedef __attribute__((ext_vector_type(8)))  _Float16 v8h;
typedef __attribute__((ext_vector_type(8)))  float    v8f;

#define BATCH   2
#define NPTS    1024
#define DMODEL  128
#define NHEAD   8
#define DKH     16
#define NLAYER  4
#define KGCN    16
#define KLOC    8
#define NCOARSE 512
#define LEPS    1e-5f

// ---------------------------------------------------------------------------
// Pack kernels: build f16 operands with K padded to a multiple of 32 so the
// GEMM runs with zero bounds checks and pure vector loads.
//   Ah[m][kp]  row-major f16           -> A-frag = 1x aligned v16h load
//   Wt[n][kp]  transposed f16          -> B-frag = 1x aligned v16h load
// ---------------------------------------------------------------------------
__global__ void pack_a_kernel(const float* __restrict__ A, _Float16* __restrict__ Ah,
                              int M, int K, int Kp) {
  int t = blockIdx.x * blockDim.x + threadIdx.x;
  if (t >= M * Kp) return;
  int k = t % Kp, m = t / Kp;
  Ah[t] = (k < K) ? (_Float16)A[(size_t)m * K + k] : (_Float16)0.0f;
}

__global__ void pack_wt_kernel(const float* __restrict__ W, _Float16* __restrict__ Wt,
                               int K, int N, int Kp) {
  int t = blockIdx.x * blockDim.x + threadIdx.x;
  if (t >= N * Kp) return;
  int k = t % Kp, n = t / Kp;
  Wt[t] = (k < K) ? (_Float16)W[(size_t)k * N + n] : (_Float16)0.0f;
}

// Q/K packed [b][h][n][16] f16 (Q pre-scaled); V packed transposed [b][h][dk][1024]
__global__ void pack_qk_kernel(const float* __restrict__ src, _Float16* __restrict__ dst,
                               float scale) {
  int t = blockIdx.x * blockDim.x + threadIdx.x;
  if (t >= BATCH * NHEAD * NPTS * DKH) return;
  int dk = t & 15, n = (t >> 4) & (NPTS - 1), h = (t >> 14) & 7, b = t >> 17;
  dst[t] = (_Float16)(src[((size_t)(b * NPTS + n)) * DMODEL + h * DKH + dk] * scale);
}

__global__ void pack_vt_kernel(const float* __restrict__ src, _Float16* __restrict__ dst) {
  int t = blockIdx.x * blockDim.x + threadIdx.x;
  if (t >= BATCH * NHEAD * DKH * NPTS) return;
  int n = t & (NPTS - 1), dk = (t >> 10) & 15, h = (t >> 14) & 7, b = t >> 17;
  dst[t] = (_Float16)src[((size_t)(b * NPTS + n)) * DMODEL + h * DKH + dk];
}

// ---------------------------------------------------------------------------
// GEMM: C[M,N] = act(A @ W + bias). One wave per 32x32 tile, 4 WMMAs per
// K-step. Requires M%32==0, N%32==0, Kp%32==0 (guaranteed by packing).
// A and B fragments both use contiguous k = 16*(lane/16) + s.
// ---------------------------------------------------------------------------
__global__ void gemm_wmma_kernel(const _Float16* __restrict__ Ah,
                                 const _Float16* __restrict__ Wt,
                                 const float* __restrict__ bias,
                                 float* __restrict__ C,
                                 int M, int N, int Kp, int act) {
  const int lane  = threadIdx.x & 31;
  const int nloc  = lane & 15;
  const int tn    = blockIdx.x * 32;
  const int tm    = blockIdx.y * 32;
  const int koffA = (lane < 16) ? 0 : 8;
  const int koffB = (lane < 16) ? 0 : 16;

  const _Float16* a0p = Ah + (size_t)(tm + nloc) * Kp;
  const _Float16* a1p = Ah + (size_t)(tm + 16 + nloc) * Kp;
  const _Float16* b0p = Wt + (size_t)(tn + nloc) * Kp;
  const _Float16* b1p = Wt + (size_t)(tn + 16 + nloc) * Kp;

  v8f acc00 = {0.f,0.f,0.f,0.f,0.f,0.f,0.f,0.f};
  v8f acc01 = acc00, acc10 = acc00, acc11 = acc00;

  for (int k0 = 0; k0 < Kp; k0 += 32) {
    if (k0 + 32 < Kp) {   // gfx1250 global_prefetch_b8 for the next K tile
      __builtin_prefetch(a0p + k0 + 32, 0, 1);
      __builtin_prefetch(b0p + k0 + 32, 0, 1);
    }
    v16h A0 = *(const v16h*)(a0p + k0 + koffB);
    v16h A1 = *(const v16h*)(a1p + k0 + koffB);
    v16h B0 = *(const v16h*)(b0p + k0 + koffB);
    v16h B1 = *(const v16h*)(b1p + k0 + koffB);
    acc00 = __builtin_amdgcn_wmma_f32_16x16x32_f16(false, A0, false, B0, (short)0, acc00, false, false);
    acc01 = __builtin_amdgcn_wmma_f32_16x16x32_f16(false, A0, false, B1, (short)0, acc01, false, false);
    acc10 = __builtin_amdgcn_wmma_f32_16x16x32_f16(false, A1, false, B0, (short)0, acc10, false, false);
    acc11 = __builtin_amdgcn_wmma_f32_16x16x32_f16(false, A1, false, B1, (short)0, acc11, false, false);
  }

  const int n0 = tn + nloc, n1 = tn + 16 + nloc;
  const float bb0 = bias ? bias[n0] : 0.0f;
  const float bb1 = bias ? bias[n1] : 0.0f;
#pragma unroll
  for (int r = 0; r < 8; ++r) {
    int m0 = tm + ((lane < 16) ? r : (8 + r));
    int m1 = m0 + 16;
    float v00 = acc00[r] + bb0, v01 = acc01[r] + bb1;
    float v10 = acc10[r] + bb0, v11 = acc11[r] + bb1;
    if (act == 1) {
      v00 = fmaxf(v00, 0.f); v01 = fmaxf(v01, 0.f);
      v10 = fmaxf(v10, 0.f); v11 = fmaxf(v11, 0.f);
    }
    C[(size_t)m0 * N + n0] = v00;
    C[(size_t)m0 * N + n1] = v01;
    C[(size_t)m1 * N + n0] = v10;
    C[(size_t)m1 * N + n1] = v11;
  }
}

// ---------------------------------------------------------------------------
// Fused attention: one wave per (b, h, 16-query tile).
// S[16x1024] via WMMA into LDS (+ alpha * <c_q,c_k>), wave softmax writes
// f16 probabilities to LDS, then P @ V via WMMA from packed f16 operands.
// LDS = 64KB f32 scores + 32KB f16 probs = 96KB (CDNA5 allows 320KB/WG).
// ---------------------------------------------------------------------------
__global__ void attn_kernel(const _Float16* __restrict__ Qh,   // [b][h][n][16], pre-scaled
                            const _Float16* __restrict__ Kh,   // [b][h][n][16]
                            const _Float16* __restrict__ Vt,   // [b][h][dk][1024]
                            const float* __restrict__ coords,
                            const float* __restrict__ alpha_arr, int layer,
                            float* __restrict__ Out) {
  extern __shared__ float sc[];                       // [16][NPTS]
  _Float16* pf = (_Float16*)(sc + 16 * NPTS);         // [16][NPTS]
  const int lane  = threadIdx.x & 31;
  const int qt    = blockIdx.x;
  const int h     = blockIdx.y;
  const int b     = blockIdx.z;
  const float alpha = alpha_arr[layer];
  const int nloc  = lane & 15;
  const int koffA = (lane < 16) ? 0 : 8;
  const int koffB = (lane < 16) ? 0 : 16;
  const size_t headQ = ((size_t)(b * NHEAD + h)) * NPTS * DKH;
  const size_t headV = ((size_t)(b * NHEAD + h)) * DKH * NPTS;

  // A fragment = Q tile (16x16 valid, K 16..31 zero padding)
  const int qrow = qt * 16 + nloc;
  v16h qv = *(const v16h*)(Qh + headQ + (size_t)qrow * DKH);
  v16h aq;
#pragma unroll
  for (int i = 0; i < 16; ++i) aq[i] = (lane < 16) ? qv[i] : (_Float16)0.0f;

  // ---- score strip S[16 x 1024] ----
  for (int j = 0; j < NPTS / 16; ++j) {
    const int key = j * 16 + nloc;
    v16h kv = *(const v16h*)(Kh + headQ + (size_t)key * DKH);  // 32B load, all lanes
    v16h bk;
#pragma unroll
    for (int i = 0; i < 16; ++i) bk[i] = (lane < 16) ? kv[i] : (_Float16)0.0f;
    v8f cs = {0.f,0.f,0.f,0.f,0.f,0.f,0.f,0.f};
    cs = __builtin_amdgcn_wmma_f32_16x16x32_f16(false, aq, false, bk, (short)0, cs, false, false);
    const float* ck = &coords[((size_t)(b * NPTS + key)) * 3];
    float kx = ck[0], ky = ck[1], kz = ck[2];
#pragma unroll
    for (int r = 0; r < 8; ++r) {
      int m = (lane < 16) ? r : (8 + r);
      const float* cq = &coords[((size_t)(b * NPTS + qt * 16 + m)) * 3];
      float g = cq[0] * kx + cq[1] * ky + cq[2] * kz;
      sc[m * NPTS + key] = cs[r] + alpha * g;
    }
  }
  __syncthreads();

  // ---- softmax each row; emit f16 probabilities ----
  for (int m = 0; m < 16; ++m) {
    float mx = -1e30f;
    for (int c = lane; c < NPTS; c += 32) mx = fmaxf(mx, sc[m * NPTS + c]);
    for (int off = 16; off; off >>= 1) mx = fmaxf(mx, __shfl_xor(mx, off, 32));
    float sum = 0.f;
    for (int c = lane; c < NPTS; c += 32) {
      float e = __expf(sc[m * NPTS + c] - mx);
      sc[m * NPTS + c] = e;
      sum += e;
    }
    for (int off = 16; off; off >>= 1) sum += __shfl_xor(sum, off, 32);
    float inv = 1.0f / sum;
    for (int c = lane; c < NPTS; c += 32)
      pf[m * NPTS + c] = (_Float16)(sc[m * NPTS + c] * inv);
  }
  __syncthreads();

  // ---- O[16x16] = P[16x1024] @ V[1024x16] ----
  v8f acc = {0.f,0.f,0.f,0.f,0.f,0.f,0.f,0.f};
  for (int kt = 0; kt < NPTS / 32; ++kt) {
    v16h ap = *(const v16h*)(pf + nloc * NPTS + kt * 32 + koffB);
    v16h bv = *(const v16h*)(Vt + headV + (size_t)nloc * NPTS + kt * 32 + koffB);
    acc = __builtin_amdgcn_wmma_f32_16x16x32_f16(false, ap, false, bv, (short)0, acc, false, false);
  }
#pragma unroll
  for (int r = 0; r < 8; ++r) {
    int m = (lane < 16) ? r : (8 + r);
    Out[((size_t)(b * NPTS + qt * 16 + m)) * DMODEL + h * DKH + nloc] = acc[r];
  }
}

// ---------------------------------------------------------------------------
// LayerNorm(x + y) * g + b ; one wave per row of 128
// ---------------------------------------------------------------------------
__global__ void ln_add_kernel(const float* __restrict__ x, const float* __restrict__ y,
                              const float* __restrict__ g, const float* __restrict__ bta,
                              float* __restrict__ out) {
  const int row  = blockIdx.x;
  const int lane = threadIdx.x & 31;
  float vals[4];
  float s = 0.f;
#pragma unroll
  for (int i = 0; i < 4; ++i) {
    int c = lane + i * 32;
    vals[i] = x[(size_t)row * DMODEL + c] + y[(size_t)row * DMODEL + c];
    s += vals[i];
  }
  for (int off = 16; off; off >>= 1) s += __shfl_xor(s, off, 32);
  float mean = s * (1.0f / DMODEL);
  float vs = 0.f;
#pragma unroll
  for (int i = 0; i < 4; ++i) { float d = vals[i] - mean; vs += d * d; }
  for (int off = 16; off; off >>= 1) vs += __shfl_xor(vs, off, 32);
  float inv = rsqrtf(vs * (1.0f / DMODEL) + LEPS);
#pragma unroll
  for (int i = 0; i < 4; ++i) {
    int c = lane + i * 32;
    out[(size_t)row * DMODEL + c] = (vals[i] - mean) * inv * g[c] + bta[c];
  }
}

// kNN (exclude self), k=16
__global__ void knn_kernel(const float* __restrict__ coords, int* __restrict__ idx) {
  int t = blockIdx.x * blockDim.x + threadIdx.x;
  if (t >= BATCH * NPTS) return;
  int b = t / NPTS, i = t % NPTS;
  const float* cb = coords + (size_t)b * NPTS * 3;
  float qx = cb[i * 3], qy = cb[i * 3 + 1], qz = cb[i * 3 + 2];
  float bd[KGCN]; int bi[KGCN];
#pragma unroll
  for (int k = 0; k < KGCN; ++k) { bd[k] = 1e30f; bi[k] = 0; }
  for (int j = 0; j < NPTS; ++j) {
    if (j == i) continue;
    float dx = qx - cb[j * 3], dy = qy - cb[j * 3 + 1], dz = qz - cb[j * 3 + 2];
    float d2 = dx * dx + dy * dy + dz * dz;
    if (d2 < bd[KGCN - 1]) {
      int p = KGCN - 1;
      while (p > 0 && bd[p - 1] > d2) { bd[p] = bd[p - 1]; bi[p] = bi[p - 1]; --p; }
      bd[p] = d2; bi[p] = j;
    }
  }
  for (int k = 0; k < KGCN; ++k) idx[(size_t)t * KGCN + k] = bi[k];
}

__global__ void gcn_agg_kernel(const float* __restrict__ x, const int* __restrict__ idx,
                               float* __restrict__ out, int C) {
  int t = blockIdx.x * blockDim.x + threadIdx.x;
  if (t >= BATCH * NPTS * C) return;
  int c  = t % C;
  int bn = t / C;
  int b  = bn / NPTS;
  float s = x[(size_t)bn * C + c];
  const int* row = idx + (size_t)bn * KGCN;
  for (int k = 0; k < KGCN; ++k)
    s += x[((size_t)b * NPTS + row[k]) * C + c];
  out[(size_t)bn * C + c] = s * (1.0f / (KGCN + 1));
}

// mean pool -> 32 rows (rows >= BATCH zero-filled so decoder GEMMs need no guards)
__global__ void meanpool_kernel(const float* __restrict__ x, float* __restrict__ gc) {
  int t = blockIdx.x * blockDim.x + threadIdx.x;
  if (t >= 32 * DMODEL) return;
  int b = t / DMODEL, c = t % DMODEL;
  float s = 0.f;
  if (b < BATCH) {
    for (int n = 0; n < NPTS; ++n) s += x[((size_t)b * NPTS + n) * DMODEL + c];
    s *= (1.0f / NPTS);
  }
  gc[t] = s;
}

// top-8 nearest partial points -> mean coords
__global__ void neigh_mean_kernel(const float* __restrict__ part,
                                  const float* __restrict__ pred,
                                  float* __restrict__ nm, int Np) {
  int t = blockIdx.x * blockDim.x + threadIdx.x;
  if (t >= BATCH * Np) return;
  int b = t / Np;
  const float* pb = part + (size_t)b * NPTS * 3;
  float qx = pred[(size_t)t * 3], qy = pred[(size_t)t * 3 + 1], qz = pred[(size_t)t * 3 + 2];
  float bd[KLOC]; int bi[KLOC];
#pragma unroll
  for (int k = 0; k < KLOC; ++k) { bd[k] = 1e30f; bi[k] = 0; }
  for (int j = 0; j < NPTS; ++j) {
    float dx = qx - pb[j * 3], dy = qy - pb[j * 3 + 1], dz = qz - pb[j * 3 + 2];
    float d2 = dx * dx + dy * dy + dz * dz;
    if (d2 < bd[KLOC - 1]) {
      int p = KLOC - 1;
      while (p > 0 && bd[p - 1] > d2) { bd[p] = bd[p - 1]; bi[p] = bi[p - 1]; --p; }
      bd[p] = d2; bi[p] = j;
    }
  }
  float sx = 0.f, sy = 0.f, sz = 0.f;
  for (int k = 0; k < KLOC; ++k) {
    sx += pb[bi[k] * 3]; sy += pb[bi[k] * 3 + 1]; sz += pb[bi[k] * 3 + 2];
  }
  nm[(size_t)t * 3]     = sx * (1.0f / KLOC);
  nm[(size_t)t * 3 + 1] = sy * (1.0f / KLOC);
  nm[(size_t)t * 3 + 2] = sz * (1.0f / KLOC);
}

__global__ void comb_kernel(const float* __restrict__ pred, const float* __restrict__ nm,
                            float* __restrict__ comb, int Np) {
  int t = blockIdx.x * blockDim.x + threadIdx.x;
  if (t >= BATCH * Np) return;
#pragma unroll
  for (int p = 0; p < 3; ++p) {
    comb[(size_t)t * 6 + p]     = pred[(size_t)t * 3 + p];
    comb[(size_t)t * 6 + 3 + p] = nm[(size_t)t * 3 + p];
  }
}

__global__ void expand_bias_kernel(const float* __restrict__ db, float* __restrict__ b256) {
  int t = threadIdx.x;
  if (t < 256) b256[t] = db[t >> 1];
}

// child[b, n*2+t, p] = pred[b,n,p] + sum_o h[bn, o*2+t] * cw[p,o] + cb[p]
__global__ void child_kernel(const float* __restrict__ h, const float* __restrict__ pred,
                             const float* __restrict__ cw, const float* __restrict__ cb,
                             float* __restrict__ out, int Np) {
  int t = blockIdx.x * blockDim.x + threadIdx.x;
  if (t >= BATCH * Np * 2) return;
  int tt = t & 1;
  int bn = t >> 1;
  int b = bn / Np, n = bn % Np;
  const float* hr = h + (size_t)bn * 256;
  float* op = out + ((size_t)b * (Np * 2) + n * 2 + tt) * 3;
#pragma unroll
  for (int p = 0; p < 3; ++p) {
    float s = cb[p];
    for (int o = 0; o < 128; ++o) s += hr[o * 2 + tt] * cw[p * 128 + o];
    op[p] = pred[(size_t)bn * 3 + p] + s;
  }
}

// ---------------------------------------------------------------------------
// Host orchestration
// ---------------------------------------------------------------------------
static inline void gemm_packed(const float* A, const float* W, const float* bias,
                               float* C, int M, int N, int K, int act,
                               _Float16* ahS, _Float16* wtS, hipStream_t s) {
  int Kp = (K + 31) & ~31;
  int ta = M * Kp, tw = N * Kp;
  pack_a_kernel<<<(ta + 255) / 256, 256, 0, s>>>(A, ahS, M, K, Kp);
  pack_wt_kernel<<<(tw + 255) / 256, 256, 0, s>>>(W, wtS, K, N, Kp);
  gemm_wmma_kernel<<<dim3(N / 32, M / 32), 32, 0, s>>>(ahS, wtS, bias, C, M, N, Kp, act);
}

extern "C" void kernel_launch(void* const* d_in, const int* in_sizes, int n_in,
                              void* d_out, int out_size, void* d_ws, size_t ws_size,
                              hipStream_t stream) {
  const float* coords  = (const float*)d_in[0];
  const float* gcn_w0  = (const float*)d_in[1];
  const float* gcn_b0  = (const float*)d_in[2];
  const float* gcn_w1  = (const float*)d_in[3];
  const float* gcn_b1  = (const float*)d_in[4];
  const float* enc_fw  = (const float*)d_in[5];
  const float* enc_fb  = (const float*)d_in[6];
  const float* t_wq    = (const float*)d_in[7];
  const float* t_bq    = (const float*)d_in[8];
  const float* t_wk    = (const float*)d_in[9];
  const float* t_bk    = (const float*)d_in[10];
  const float* t_wv    = (const float*)d_in[11];
  const float* t_bv    = (const float*)d_in[12];
  const float* t_alpha = (const float*)d_in[13];
  const float* t_f1w   = (const float*)d_in[14];
  const float* t_f1b   = (const float*)d_in[15];
  const float* t_f2w   = (const float*)d_in[16];
  const float* t_f2b   = (const float*)d_in[17];
  const float* t_ln1g  = (const float*)d_in[18];
  const float* t_ln1b  = (const float*)d_in[19];
  const float* t_ln2g  = (const float*)d_in[20];
  const float* t_ln2b  = (const float*)d_in[21];
  const float* dec_w1  = (const float*)d_in[22];
  const float* dec_b1  = (const float*)d_in[23];
  const float* dec_w2  = (const float*)d_in[24];
  const float* dec_b2  = (const float*)d_in[25];
  const float* st_sw1  = (const float*)d_in[26];
  const float* st_sb1  = (const float*)d_in[27];
  const float* st_sw2  = (const float*)d_in[28];
  const float* st_sb2  = (const float*)d_in[29];
  const float* st_dw   = (const float*)d_in[30];
  const float* st_db   = (const float*)d_in[31];
  const float* st_cw   = (const float*)d_in[32];
  const float* st_cb   = (const float*)d_in[33];
  (void)in_sizes; (void)n_in; (void)out_size; (void)ws_size;

  char* ws = (char*)d_ws;
  size_t off = 0;
  auto alloc = [&](size_t bytes) -> void* {
    void* p = ws + off;
    off = (off + bytes + 255) & ~(size_t)255;
    return p;
  };

  const int ROWS = BATCH * NPTS;              // 2048
  int*      idx    = (int*)      alloc((size_t)ROWS * KGCN * 4);
  float*    agg3   = (float*)    alloc((size_t)ROWS * 3 * 4);
  float*    x64    = (float*)    alloc((size_t)ROWS * 64 * 4);
  float*    agg64  = (float*)    alloc((size_t)ROWS * 64 * 4);
  float*    xA     = (float*)    alloc((size_t)ROWS * DMODEL * 4);
  float*    xB     = (float*)    alloc((size_t)ROWS * DMODEL * 4);
  float*    qb     = (float*)    alloc((size_t)ROWS * DMODEL * 4);
  float*    kb     = (float*)    alloc((size_t)ROWS * DMODEL * 4);
  float*    vb     = (float*)    alloc((size_t)ROWS * DMODEL * 4);
  float*    attnb  = (float*)    alloc((size_t)ROWS * DMODEL * 4);
  float*    ffb    = (float*)    alloc((size_t)ROWS * 4 * DMODEL * 4);
  float*    gc     = (float*)    alloc((size_t)32 * DMODEL * 4);
  float*    dech   = (float*)    alloc((size_t)32 * DMODEL * 4);
  float*    coarse = (float*)    alloc((size_t)32 * 3 * NCOARSE * 4);
  float*    ptsA   = (float*)    alloc((size_t)BATCH * 1024 * 3 * 4);
  float*    ptsB   = (float*)    alloc((size_t)BATCH * 2048 * 3 * 4);
  float*    nm     = (float*)    alloc((size_t)BATCH * 2048 * 3 * 4);
  float*    comb   = (float*)    alloc((size_t)BATCH * 2048 * 6 * 4);
  float*    t1     = (float*)    alloc((size_t)BATCH * 2048 * 128 * 4);
  float*    seed   = (float*)    alloc((size_t)BATCH * 2048 * 128 * 4);
  float*    hb     = (float*)    alloc((size_t)BATCH * 2048 * 256 * 4);
  float*    b256   = (float*)    alloc(256 * 4);
  _Float16* qh     = (_Float16*) alloc((size_t)BATCH * NHEAD * NPTS * DKH * 2);
  _Float16* kh     = (_Float16*) alloc((size_t)BATCH * NHEAD * NPTS * DKH * 2);
  _Float16* vth    = (_Float16*) alloc((size_t)BATCH * NHEAD * DKH * NPTS * 2);
  _Float16* ahS    = (_Float16*) alloc((size_t)ROWS * 512 * 2);       // max M*Kp
  _Float16* wtS    = (_Float16*) alloc((size_t)1536 * 128 * 2);       // max N*Kp

  // ---- GraphEncoder ----
  knn_kernel<<<(ROWS + 255) / 256, 256, 0, stream>>>(coords, idx);
  gcn_agg_kernel<<<(ROWS * 3 + 255) / 256, 256, 0, stream>>>(coords, idx, agg3, 3);
  gemm_packed(agg3, gcn_w0, gcn_b0, x64, ROWS, 64, 3, 1, ahS, wtS, stream);
  gcn_agg_kernel<<<(ROWS * 64 + 255) / 256, 256, 0, stream>>>(x64, idx, agg64, 64);
  gemm_packed(agg64, gcn_w1, gcn_b1, xB, ROWS, 128, 64, 1, ahS, wtS, stream);
  gemm_packed(xB, enc_fw, enc_fb, xA, ROWS, 128, 128, 0, ahS, wtS, stream);

  // ---- GeomMultiTokenTransformer ----
  const int QKVE = BATCH * NHEAD * NPTS * DKH;   // 262144
  for (int i = 0; i < NLAYER; ++i) {
    gemm_packed(xA, t_wq + (size_t)i * 128 * 128, t_bq + (size_t)i * 128, qb, ROWS, 128, 128, 0, ahS, wtS, stream);
    gemm_packed(xA, t_wk + (size_t)i * 128 * 128, t_bk + (size_t)i * 128, kb, ROWS, 128, 128, 0, ahS, wtS, stream);
    gemm_packed(xA, t_wv + (size_t)i * 128 * 128, t_bv + (size_t)i * 128, vb, ROWS, 128, 128, 0, ahS, wtS, stream);
    pack_qk_kernel<<<(QKVE + 255) / 256, 256, 0, stream>>>(qb, qh, 0.25f);
    pack_qk_kernel<<<(QKVE + 255) / 256, 256, 0, stream>>>(kb, kh, 1.0f);
    pack_vt_kernel<<<(QKVE + 255) / 256, 256, 0, stream>>>(vb, vth);
    attn_kernel<<<dim3(NPTS / 16, NHEAD, BATCH), 32, 16 * NPTS * 4 + 16 * NPTS * 2, stream>>>(
        qh, kh, vth, coords, t_alpha, i, attnb);
    ln_add_kernel<<<ROWS, 32, 0, stream>>>(xA, attnb, t_ln1g + (size_t)i * 128,
                                           t_ln1b + (size_t)i * 128, xB);
    gemm_packed(xB, t_f1w + (size_t)i * 128 * 512, t_f1b + (size_t)i * 512, ffb, ROWS, 512, 128, 1, ahS, wtS, stream);
    gemm_packed(ffb, t_f2w + (size_t)i * 512 * 128, t_f2b + (size_t)i * 128, attnb, ROWS, 128, 512, 0, ahS, wtS, stream);
    ln_add_kernel<<<ROWS, 32, 0, stream>>>(xB, attnb, t_ln2g + (size_t)i * 128,
                                           t_ln2b + (size_t)i * 128, xA);
  }

  // ---- Snowflake3StageDecoder: coarse (M padded to 32) ----
  meanpool_kernel<<<(32 * DMODEL + 255) / 256, 256, 0, stream>>>(xA, gc);
  gemm_packed(gc, dec_w1, dec_b1, dech, 32, 128, 128, 1, ahS, wtS, stream);
  gemm_packed(dech, dec_w2, dec_b2, coarse, 32, 3 * NCOARSE, 128, 0, ahS, wtS, stream);

  // ---- 3 refinement stages: 512 -> 1024 -> 2048 -> 4096 ----
  float* pts = coarse;
  float* outs[3] = { ptsA, ptsB, (float*)d_out };
  for (int s = 0; s < 3; ++s) {
    int Np = NCOARSE << s;
    int rows = BATCH * Np;
    neigh_mean_kernel<<<(rows + 127) / 128, 128, 0, stream>>>(coords, pts, nm, Np);
    comb_kernel<<<(rows + 127) / 128, 128, 0, stream>>>(pts, nm, comb, Np);
    gemm_packed(comb, st_sw1 + (size_t)s * 6 * 128,   st_sb1 + (size_t)s * 128, t1,   rows, 128, 6,   1, ahS, wtS, stream);
    gemm_packed(t1,   st_sw2 + (size_t)s * 128 * 128, st_sb2 + (size_t)s * 128, seed, rows, 128, 128, 0, ahS, wtS, stream);
    expand_bias_kernel<<<1, 256, 0, stream>>>(st_db + (size_t)s * 128, b256);
    gemm_packed(seed, st_dw + (size_t)s * 128 * 256, b256, hb, rows, 256, 128, 1, ahS, wtS, stream);
    child_kernel<<<(rows * 2 + 127) / 128, 128, 0, stream>>>(
        hb, pts, st_cw + (size_t)s * 3 * 128, st_cb + (size_t)s * 3, outs[s], Np);
    pts = outs[s];
  }
}
